// CKANLayer_61074434949894
// MI455X (gfx1250) — matrix operand order repaired and checked by reading, not compiled, not run on hardware
//
#include <hip/hip_runtime.h>
#include <hip/hip_bf16.h>
#include <stddef.h>
#include <stdint.h>

// ---------------------------------------------------------------------------
// KAN conv layer as fused bf16 WMMA GEMM for gfx1250 (MI455X).
//
// Per-pixel nonlinearity [silu(v), B0(v)..B10(v)] depends only on the pixel
// value (all knot rows identical), so the layer collapses to a 3x3 conv with
// 64*16 effective in-channels (feature dim padded 12->16):
//   out[co,n,h,w] = sum_{c,tap,f} W2[co][(c*9+tap)*16+f] * F[n][c][h+ki][w+kj][f]
// GEMM M=128, N=8192, K=9216 -> v_wmma_f32_16x16x32_bf16, f32 accumulate.
// F tiles staged to LDS by the Tensor Data Mover (TENSOR_LOAD_TO_LDS, one
// descriptor per 68KB chunk, TENSORcnt) with a double-buffered pipeline;
// falls back to GLOBAL_LOAD_ASYNC_TO_LDS_B128 (ASYNCcnt), then to plain
// load+ds_store, depending on available builtins.
// ---------------------------------------------------------------------------

typedef __bf16  v16bf __attribute__((ext_vector_type(16)));
typedef __bf16  v8bf  __attribute__((ext_vector_type(8)));
typedef float   v8f   __attribute__((ext_vector_type(8)));

// Types matching the async-LDS builtin signature:
//   void __builtin_amdgcn_global_load_async_to_lds_b128(v4i AS1*, v4i AS3*, imm, imm)
typedef int v4i_ __attribute__((vector_size(16)));
typedef __attribute__((address_space(1))) v4i_ as1_v4i;
typedef __attribute__((address_space(3))) v4i_ as3_v4i;

// TDM descriptor groups (CDNA5 ISA):
typedef unsigned int v4u_ __attribute__((vector_size(16)));  // D# group 0
typedef int          v8i_ __attribute__((vector_size(32)));  // D# group 1

#define IN_CH   64
#define OUT_CH  128
#define KS      3
#define HW      32         // spatial
#define HP      34         // padded spatial
#define FEAT    16         // padded feature dim (silu + 11 basis + 4 zeros)
#define KTOT    (IN_CH * KS * KS * FEAT)   // 9216
#define SIZE_S  (IN_CH * KS * KS)          // 576
#define NCOEF   11

#define CH_STRIDE   (HP * HP * FEAT)       // 18496 elems per channel image
#define ROW_STRIDE  (HP * FEAT)            // 544 elems per padded row
#define TILE_ELEMS  (16 * 4 * HP * FEAT)   // 34816 elems per LDS chunk (68KB)
#define TILE_UNITS  (16 * 4 * HP * 2)      // 4352 x 16B units per chunk
#define TILE_BYTES  (TILE_ELEMS * 2)       // 69632 B

#if __has_builtin(__builtin_amdgcn_tensor_load_to_lds) && \
    __has_builtin(__builtin_amdgcn_s_wait_tensorcnt)
#define USE_TDM 1
#else
#define USE_TDM 0
#endif

#if __has_builtin(__builtin_amdgcn_global_load_async_to_lds_b128) && \
    __has_builtin(__builtin_amdgcn_s_wait_asynccnt)
#define USE_ASYNC_LDS 1
#else
#define USE_ASYNC_LDS 0
#endif

// ---------------------------------------------------------------------------
// Kernel 1: fuse weights -> W2[OUT_CH][KTOT] bf16, k = (c*9 + tap)*16 + f
// ---------------------------------------------------------------------------
__global__ __launch_bounds__(256) void prep_weights(
    const float* __restrict__ coeff,     // (128,576,11)
    const float* __restrict__ base_w,    // (128,576)
    const float* __restrict__ spline_w,  // (128,576)
    __bf16* __restrict__ W2)             // (128,9216)
{
    int idx = blockIdx.x * 256 + threadIdx.x;      // 128*9216 = 1179648 total
    int co  = idx / KTOT;
    int k   = idx - co * KTOT;
    int c   = k / (KS * KS * FEAT);                // /144
    int rem = k - c * (KS * KS * FEAT);
    int tap = rem / FEAT;
    int f   = rem - tap * FEAT;
    int s   = c * (KS * KS) + tap;

    float v = 0.0f;
    if (f == 0) {
        v = base_w[co * SIZE_S + s];
    } else if (f <= NCOEF) {
        v = spline_w[co * SIZE_S + s] * coeff[(co * SIZE_S + s) * NCOEF + (f - 1)];
    }
    W2[idx] = (__bf16)v;
}

// ---------------------------------------------------------------------------
// Kernel 2: per padded pixel compute [silu(u), B0..B10(u), 0,0,0,0] bf16.
// F layout: [n][c][hp(34)][wp(34)][16].  Border uses u = 0 (reference pads
// BEFORE the basis, so the border carries B(0) != 0).
// ---------------------------------------------------------------------------
__global__ __launch_bounds__(256) void prep_features(
    const float* __restrict__ x,       // (8,64,32,32)
    const float* __restrict__ knots,   // (576,9) -- all rows identical
    __bf16* __restrict__ F)            // (8,64,34,34,16)
{
    int idx = blockIdx.x * 256 + threadIdx.x;      // 8*64*34*34 = 591872
    int wp = idx % HP;
    int t1 = idx / HP;
    int hp = t1 % HP;
    int t2 = t1 / HP;
    int c  = t2 % IN_CH;
    int n  = t2 / IN_CH;

    float u = 0.0f;
    if (hp >= 1 && hp <= HW && wp >= 1 && wp <= HW) {
        u = x[(((size_t)n * IN_CH + c) * HW + (hp - 1)) * HW + (wp - 1)];
    }

    float k0 = knots[0];
    float hs = (knots[8] - knots[0]) * 0.125f;     // uniform step

    // degree-0 indicators on 14 intervals of extended grid g_i = k0 + hs*(i-3)
    float B[14];
#pragma unroll
    for (int j = 0; j < 14; ++j) {
        float gj  = k0 + hs * (float)(j - 3);
        float gj1 = gj + hs;
        B[j] = (u >= gj && u < gj1) ? 1.0f : 0.0f;
    }
    // Cox-de-Boor recursion up to degree 3 (uniform knots: both denoms = d*hs)
#pragma unroll
    for (int d = 1; d <= 3; ++d) {
        float inv = 1.0f / (hs * (float)d);
#pragma unroll 13
        for (int j = 0; j < 13; ++j) {
            if (j < 14 - d) {
                float gj   = k0 + hs * (float)(j - 3);
                float gjd1 = gj + hs * (float)(d + 1);
                B[j] = (u - gj) * inv * B[j] + (gjd1 - u) * inv * B[j + 1];
            }
        }
    }

    float silu = u / (1.0f + __expf(-u));

    union { __bf16 b[16]; uint4 q[2]; } pack;
    pack.b[0] = (__bf16)silu;
#pragma unroll
    for (int j = 0; j < NCOEF; ++j) pack.b[1 + j] = (__bf16)B[j];
#pragma unroll
    for (int j = NCOEF + 1; j < FEAT; ++j) pack.b[j] = (__bf16)0.0f;

    uint4* dst = (uint4*)(F + (size_t)idx * FEAT);
    dst[0] = pack.q[0];
    dst[1] = pack.q[1];
}

#if USE_TDM
// ---------------------------------------------------------------------------
// Issue one TDM descriptor: 2D tile, dim0 = 4 rows * 544 elems contiguous
// per channel (1088 u32), dim1 = 16 channels, stride 9248 u32 (36992 B).
// D# fields per CDNA5 ISA.
// ---------------------------------------------------------------------------
__device__ __forceinline__ void tdm_stage(const __bf16* gsrc, unsigned lds_byte_off)
{
    unsigned long long ga = (unsigned long long)(uintptr_t)gsrc;
    v4u_ g0;
    g0[0] = 1u;                                              // count=1, valid
    g0[1] = lds_byte_off;                                    // lds_addr
    g0[2] = (unsigned)ga;                                    // global_addr[31:0]
    g0[3] = ((unsigned)(ga >> 32) & 0x01FFFFFFu)             // global_addr[56:32]
          | 0x80000000u;                                     // type=2 (image)
    v8i_ g1;
    g1[0] = 0x00020000;          // workgroup_mask=0, data_size=2 (4 bytes)
    g1[1] = (int)(1088u << 16);  // tensor_dim0 = 1088 u32  (bits 79:48, lo16)
    g1[2] = (int)(16u << 16);    // tensor_dim1 = 16        (bits 111:80, lo16)
    g1[3] = (int)(1088u << 16);  // tile_dim0   = 1088      (bits 127:112)
    g1[4] = 16;                  // tile_dim1   = 16        (bits 143:128)
    g1[5] = 9248;                // tensor_dim0_stride u32  (bits 207:160, lo32)
    g1[6] = 0;
    g1[7] = 0;
    v4i_ g2 = {0, 0, 0, 0};      // dims 2/3 unused (2D tensor)
    v4i_ g3 = {0, 0, 0, 0};
#if __clang_major__ >= 23
    v8i_ g4 = {0, 0, 0, 0, 0, 0, 0, 0};
    __builtin_amdgcn_tensor_load_to_lds(g0, g1, g2, g3, g4, 0);
#else
    __builtin_amdgcn_tensor_load_to_lds(g0, g1, g2, g3, 0);
#endif
}
#endif

// ---------------------------------------------------------------------------
// Per-lane staging fallback (async-to-LDS or plain copy).
// Each of the 256 threads moves exactly 17 x 16B.
// ---------------------------------------------------------------------------
__device__ __forceinline__ void stage_chunk(const __bf16* __restrict__ Fchunk,
                                            __bf16* __restrict__ dst, int tid)
{
#pragma unroll 1
    for (int q = tid; q < TILE_UNITS; q += 256) {
        const size_t gofs = (size_t)q * 8;   // chunk is fully contiguous per
                                             // channel; channels back-to-back in
                                             // tile order c,row,wp,f
        const int cell    = q >> 1;
        const int c_local = cell / (4 * HP);
        const int rem     = cell - c_local * (4 * HP);
        const size_t gsrc = (size_t)c_local * CH_STRIDE + (size_t)rem * FEAT * 2
                          + (size_t)(q & 1) * 8;
#if USE_ASYNC_LDS
        __builtin_amdgcn_global_load_async_to_lds_b128(
            (as1_v4i*)(Fchunk + gsrc),
            (as3_v4i*)(dst + gofs),
            0, 0);
#else
        *(uint4*)(dst + gofs) = *(const uint4*)(Fchunk + gsrc);
#endif
    }
}

// ---------------------------------------------------------------------------
// Kernel 3: implicit-GEMM conv with WMMA + double-buffered TDM pipeline.
// Grid: 128 WGs x 256 thr (8 waves). WG owns (n, rows h0,h0+1, all 32 cols).
// Wave w owns out-channels [16w,16w+16) x 4 accumulators (16x16 each).
//
// WMMA bf16 fragment maps (wave32):
//   A 16x32 (MxK): lane m = lane&15, g = lane>>4;
//                  half e -> K = 16g + e (chan c+g, feat e)
//   B 32x16 (KxN): lane n = lane&15, g = lane>>4; half e -> K = 16g + e
//   C/D 16x16 f32: lane n = lane&15; vgpr r -> M = r + 8*(lane>>4)
// K-chunk (32) = two channels' 16-feat groups at one tap.
// LDS chunk layout: [c_local(16)][row(4)][wp(34)][f(16)] bf16 (rows h0..h0+3
// are contiguous in F, so chunk source is 16 lines of 4352 contiguous bytes).
// ---------------------------------------------------------------------------
__global__ __launch_bounds__(256) void kan_conv_wmma(
    const __bf16* __restrict__ W2,   // (128,9216)
    const __bf16* __restrict__ F,    // (8,64,34,34,16)
    float* __restrict__ out)         // (8,128,32,32)
{
    extern __shared__ __align__(32) __bf16 smem[];   // 2 x 68KB chunks

    const int tid  = threadIdx.x;
    const int wave = tid >> 5;
    const int lane = tid & 31;
    const int lN   = lane & 15;
    const int g    = lane >> 4;

    const int p0 = blockIdx.x * 64;
    const int n  = p0 >> 10;                 // batch index
    const int h0 = (p0 >> 5) & 31;           // even output row

    v8f acc[4] = {v8f{0.f,0.f,0.f,0.f,0.f,0.f,0.f,0.f},
                  v8f{0.f,0.f,0.f,0.f,0.f,0.f,0.f,0.f},
                  v8f{0.f,0.f,0.f,0.f,0.f,0.f,0.f,0.f},
                  v8f{0.f,0.f,0.f,0.f,0.f,0.f,0.f,0.f}};

    const __bf16* wrow = W2 + (size_t)(wave * 16 + lN) * KTOT;
    // Global base of this WG's F window (rows h0..h0+3, all cols/feats)
    const __bf16* Fwin = F + ((size_t)n * IN_CH * HP + h0) * ROW_STRIDE;

    // Prologue: stage chunk 0 into buffer 0
#if USE_TDM
    if (tid < 32) tdm_stage(Fwin, 0);
#else
    stage_chunk(Fwin, smem, tid);
#endif

    for (int cb = 0; cb < 4; ++cb) {
        __syncthreads();   // spare buffer no longer being read by any wave
        if (cb < 3) {
            const __bf16* src = Fwin + (size_t)(cb + 1) * 16 * CH_STRIDE;
#if USE_TDM
            if (tid < 32) tdm_stage(src, ((cb + 1) & 1) * TILE_BYTES);
            __builtin_amdgcn_s_wait_tensorcnt(1);  // retire chunk cb's DMA
#else
            stage_chunk(src, smem + ((cb + 1) & 1) * TILE_ELEMS, tid);
#if USE_ASYNC_LDS
            __builtin_amdgcn_s_wait_asynccnt(17);  // retire chunk cb's 17
#endif
#endif
        } else {
#if USE_TDM
            __builtin_amdgcn_s_wait_tensorcnt(0);  // last chunk fully landed
#elif USE_ASYNC_LDS
            __builtin_amdgcn_s_wait_asynccnt(0);
#endif
        }
        __syncthreads();   // all data for chunk cb visible to all waves

        const __bf16* sb = smem + (cb & 1) * TILE_ELEMS;
        const int c_base = cb * 16;

        for (int tap = 0; tap < 9; ++tap) {
            const int ki = tap / 3;
            const int kj = tap - ki * 3;
            for (int cc = 0; cc < 8; ++cc) {
                const int c  = c_base + 2 * cc;          // absolute channel
                const int cl = 2 * cc + g;               // LDS channel (this half)
                // A fragment: halves[0..15]=chan c+g, feats 0..15
                union { v16bf v; v8bf h[2]; } a;
                a.h[0] = *(const v8bf*)(wrow + ((c + g) * 9 + tap) * FEAT);
                a.h[1] = *(const v8bf*)(wrow + ((c + g) * 9 + tap) * FEAT + 8);
#pragma unroll
                for (int t = 0; t < 4; ++t) {
                    const int r  = t >> 1;               // tile row 0/1
                    const int wb = (t & 1) << 4;         // tile col base 0/16
                    const int wp = wb + lN + kj;         // padded col 0..33
                    const v16bf b =
                        *(const v16bf*)(sb + ((cl * 4 + (r + ki)) * HP + wp) * FEAT);
                    acc[t] = __builtin_amdgcn_wmma_f32_16x16x32_bf16(
                        false, a.v, false, b, (short)0, acc[t], false, false);
                }
            }
        }
    }

    // Write out: out[n][co][h][w]
#pragma unroll
    for (int t = 0; t < 4; ++t) {
        const int h = h0 + (t >> 1);
        const int w = ((t & 1) << 4) + lN;
#pragma unroll
        for (int r = 0; r < 8; ++r) {
            const int co = wave * 16 + g * 8 + r;
            out[(((size_t)n * OUT_CH + co) * HW + h) * HW + w] = acc[t][r];
        }
    }
}

// ---------------------------------------------------------------------------
extern "C" void kernel_launch(void* const* d_in, const int* in_sizes, int n_in,
                              void* d_out, int out_size, void* d_ws, size_t ws_size,
                              hipStream_t stream) {
    (void)in_sizes; (void)n_in; (void)out_size; (void)ws_size;
    const float* x        = (const float*)d_in[0];
    const float* knots    = (const float*)d_in[1];
    const float* coeff    = (const float*)d_in[2];
    const float* base_w   = (const float*)d_in[3];
    const float* spline_w = (const float*)d_in[4];
    float* out = (float*)d_out;

    // Workspace carve-out: W2 (2,359,296 B) then F (18,939,904 B) ~ 21.3 MB.
    __bf16* W2 = (__bf16*)d_ws;
    __bf16* F  = (__bf16*)((char*)d_ws + (size_t)OUT_CH * KTOT * sizeof(__bf16));

    // 128*9216 / 256 = 4608 blocks
    prep_weights<<<4608, 256, 0, stream>>>(coeff, base_w, spline_w, W2);
    // 8*64*34*34 / 256 = 2312 blocks
    prep_features<<<2312, 256, 0, stream>>>(x, knots, F);
    // 8192 positions / 64 per WG = 128 blocks; 2 x 68KB dynamic LDS
    kan_conv_wmma<<<128, 256, 2 * TILE_ELEMS * sizeof(__bf16), stream>>>(W2, F, out);
}
